// SelfAttention_33861522162367
// MI455X (gfx1250) — matrix operand order repaired and checked by reading, not compiled, not run on hardware
//
#include <hip/hip_runtime.h>
#include <hip/hip_bf16.h>

typedef __attribute__((ext_vector_type(16))) __bf16 v16bf;
typedef __attribute__((ext_vector_type(8)))  float  v8f;
typedef __attribute__((ext_vector_type(4)))  int    v4i;

union FragBF { v16bf v; uint4 q[2]; };

#define BATCH   4
#define SEQ     2048
#define DIM     1024
#define NHEADS  16
#define HD      64
#define NOUT    3072
#define MTOT    (BATCH * SEQ)
#define LDSTR   40   // bf16 elements per LDS row (32 data + 8 pad)

__device__ __forceinline__ v8f v8f_zero() {
  v8f z;
#pragma unroll
  for (int i = 0; i < 8; ++i) z[i] = 0.0f;
  return z;
}

__device__ __forceinline__ v8f wmma_bf16(v16bf a, v16bf b, v8f c) {
  return __builtin_amdgcn_wmma_f32_16x16x32_bf16(false, a, false, b,
                                                 (short)0, c, false, false);
}

// ---- CDNA5 async global->LDS copy (ASYNCcnt path), with safe fallback -----
#if defined(__has_builtin)
#if __has_builtin(__builtin_amdgcn_global_load_async_to_lds_b128)
#define HAVE_ASYNC_LDS 1
#endif
#endif

typedef __attribute__((address_space(1))) v4i* gv4i_p;   // global int4*
typedef __attribute__((address_space(3))) v4i* lv4i_p;   // LDS int4*

__device__ __forceinline__ void async_cp16(const __bf16* g, __bf16* l) {
#ifdef HAVE_ASYNC_LDS
  __builtin_amdgcn_global_load_async_to_lds_b128((gv4i_p)g, (lv4i_p)l, 0, 0);
#else
  *(uint4*)l = *(const uint4*)g;   // synchronous fallback
#endif
}

__device__ __forceinline__ void wait_async_all() {
#ifdef HAVE_ASYNC_LDS
#if __has_builtin(__builtin_amdgcn_s_wait_asynccnt)
  __builtin_amdgcn_s_wait_asynccnt(0);
#else
  asm volatile("s_wait_asynccnt 0" ::: "memory");
#endif
#endif
}

// ---------------------------------------------------------------------------
// Kernel 0a: x (f32) -> xb (bf16), elementwise, 8 elems/thread, coalesced.
// ---------------------------------------------------------------------------
__global__ __launch_bounds__(256) void cvt_x_kernel(
    const float* __restrict__ x, __bf16* __restrict__ xb)
{
  size_t i = ((size_t)blockIdx.x * 256 + threadIdx.x);
  const float4* xf = (const float4*)x;
  float4 a = xf[i * 2 + 0];
  float4 b = xf[i * 2 + 1];
  union { __bf16 h[8]; uint4 u; } o;
  o.h[0] = (__bf16)a.x; o.h[1] = (__bf16)a.y;
  o.h[2] = (__bf16)a.z; o.h[3] = (__bf16)a.w;
  o.h[4] = (__bf16)b.x; o.h[5] = (__bf16)b.y;
  o.h[6] = (__bf16)b.z; o.h[7] = (__bf16)b.w;
  ((uint4*)xb)[i] = o.u;
}

// ---------------------------------------------------------------------------
// Kernel 0b: W [k][n] (f32) -> wt [n][k] (bf16), LDS-tiled 64x64 transpose.
// ---------------------------------------------------------------------------
__global__ __launch_bounds__(256) void cvt_w_kernel(
    const float* __restrict__ w, __bf16* __restrict__ wt)
{
  __shared__ float tile[64][65];
  const int tid = threadIdx.x;
  const int nb = blockIdx.x * 64;   // along NOUT
  const int kb = blockIdx.y * 64;   // along DIM (k)
  const int tx = tid & 63, ty0 = tid >> 6;
#pragma unroll
  for (int i = 0; i < 16; ++i) {
    int ty = ty0 + i * 4;
    tile[ty][tx] = w[(size_t)(kb + ty) * NOUT + nb + tx];
  }
  __syncthreads();
  const int kx = tid & 63, ny0 = tid >> 6;
#pragma unroll
  for (int i = 0; i < 16; ++i) {
    int ny = ny0 + i * 4;
    wt[(size_t)(nb + ny) * DIM + kb + kx] = (__bf16)tile[kx][ny];
  }
}

// ---------------------------------------------------------------------------
// Kernel 1: QKV projection (bf16 WMMA, f32 accumulate), double-buffered LDS
// with async global->LDS tile staging; bias + partial RoPE epilogue writing
// q/k/v bf16 in [b, h, n, d] layout.
// ---------------------------------------------------------------------------
__global__ __launch_bounds__(256) void qkv_rope_kernel(
    const __bf16* __restrict__ xb, const __bf16* __restrict__ wt,
    const float* __restrict__ bias,
    __bf16* __restrict__ qw, __bf16* __restrict__ kw, __bf16* __restrict__ vw)
{
  __shared__ __bf16 lA[2][128 * LDSTR];   // x tile [m][k], double buffered
  __shared__ __bf16 lB[2][128 * LDSTR];   // W tile [n][k], double buffered

  const int tid  = threadIdx.x;
  const int lane = tid & 31;
  const int wid  = tid >> 5;
  const int ln   = lane & 15;
  const int hi   = lane >> 4;

  const int tile_n = blockIdx.x % (NOUT / 128);
  const int tile_m = blockIdx.x / (NOUT / 128);
  const int m0 = tile_m * 128;
  const int n0 = tile_n * 128;

  const int wm = wid & 1;    // 2 wave-rows of 64
  const int wn = wid >> 1;   // 4 wave-cols of 32

  v8f acc[4][2];
#pragma unroll
  for (int i = 0; i < 4; ++i)
#pragma unroll
    for (int j = 0; j < 2; ++j) acc[i][j] = v8f_zero();

  // async tile stage: 128 rows x 32 k per operand = 512 x 16B chunks each
  auto stage = [&](int buf, int kb) {
#pragma unroll
    for (int i = 0; i < 2; ++i) {
      int idx = tid + i * 256;
      int row = idx >> 2, c = (idx & 3) * 8;
      async_cp16(&xb[(size_t)(m0 + row) * DIM + kb + c],
                 &lA[buf][row * LDSTR + c]);
      async_cp16(&wt[(size_t)(n0 + row) * DIM + kb + c],
                 &lB[buf][row * LDSTR + c]);
    }
  };

  stage(0, 0);
  int s = 0;
  for (int kb = 0; kb < DIM; kb += 32, s ^= 1) {
    wait_async_all();
    __syncthreads();                       // buffer s fully staged, all waves
    if (kb + 32 < DIM) stage(s ^ 1, kb + 32);

    FragBF af[4], bfr[2];
#pragma unroll
    for (int i = 0; i < 4; ++i) {
      int row = wm * 64 + i * 16 + ln;
      const __bf16* p = &lA[s][row * LDSTR + hi * 8];
      af[i].q[0] = *(const uint4*)(p);
      af[i].q[1] = *(const uint4*)(p + 16);
    }
#pragma unroll
    for (int j = 0; j < 2; ++j) {
      int col = wn * 32 + j * 16 + ln;
      const __bf16* p = &lB[s][col * LDSTR + hi * 8];
      bfr[j].q[0] = *(const uint4*)(p);
      bfr[j].q[1] = *(const uint4*)(p + 16);
    }
#pragma unroll
    for (int i = 0; i < 4; ++i)
#pragma unroll
      for (int j = 0; j < 2; ++j)
        acc[i][j] = wmma_bf16(af[i].v, bfr[j].v, acc[i][j]);
  }

  // epilogue: bias + partial interleaved RoPE (first 32 of each head's 64)
  const float LOG_THETA_OVER_16 = 9.210340371976184f / 16.0f;
#pragma unroll
  for (int i = 0; i < 4; ++i) {
#pragma unroll
    for (int j = 0; j < 2; ++j) {
      int col     = n0 + wn * 32 + j * 16 + ln;
      int section = col / DIM;      // 0=q 1=k 2=v (uniform per tile)
      int c       = col % DIM;
      int head    = c / HD;
      int d       = c % HD;
      bool do_rope = (section < 2) && (d < 32);   // wave-uniform
      float bia = bias[col];
      float inv_freq = __expf(-(float)(d >> 1) * LOG_THETA_OVER_16);
      __bf16* dst = (section == 0) ? qw : (section == 1) ? kw : vw;
      int rbase = m0 + wm * 64 + i * 16 + hi * 8;
#pragma unroll
      for (int r = 0; r < 8; ++r) {
        int row = rbase + r;
        int b   = row >> 11;          // / SEQ
        int n   = row & (SEQ - 1);
        float v = acc[i][j][r] + bia;
        if (do_rope) {
          float vn  = __shfl_xor(v, 1);            // pair partner column
          float rot = (d & 1) ? vn : -vn;
          float sn, cs;
          __sincosf((float)n * inv_freq, &sn, &cs);
          v = v * cs + rot * sn;
        }
        dst[(((size_t)b * NHEADS + head) * SEQ + n) * HD + d] = (__bf16)v;
      }
    }
  }
}

// ---------------------------------------------------------------------------
// Kernel 2: flash attention. One block = 128 queries of one (b,h); each wave
// owns 16 query rows. Keys processed in chunks of 32 with bf16 WMMA for both
// Q*K^T and P*V, online softmax in f32 registers.
// ---------------------------------------------------------------------------
__global__ __launch_bounds__(256) void attn_kernel(
    const __bf16* __restrict__ qw, const __bf16* __restrict__ kw,
    const __bf16* __restrict__ vw, float* __restrict__ out)
{
  __shared__ __bf16 lVt[HD * LDSTR];          // V chunk transposed [d][key]
  __shared__ __bf16 lP[8 * 16 * LDSTR];       // per-wave P tile [row][key]

  const int tid  = threadIdx.x;
  const int lane = tid & 31;
  const int wid  = tid >> 5;
  const int ln   = lane & 15;
  const int hi   = lane >> 4;

  const int nblk = SEQ / 128;                 // 16 query blocks per (b,h)
  const int bh   = blockIdx.x / nblk;
  const int qblk = blockIdx.x % nblk;
  const int b    = bh / NHEADS;
  const int head = bh % NHEADS;
  const size_t base = (size_t)bh * SEQ * HD;

  const int q0 = qblk * 128 + wid * 16;       // this wave's first query row

  // resident Q fragments (16 rows x 64 dims = two K=32 A-fragments)
  FragBF fq[2];
  {
    const __bf16* qp = qw + base + (size_t)(q0 + ln) * HD + hi * 8;
#pragma unroll
    for (int t = 0; t < 2; ++t) {
      fq[t].q[0] = *(const uint4*)(qp + t * 32);
      fq[t].q[1] = *(const uint4*)(qp + t * 32 + 16);
    }
  }

  float m_i[8], l_i[8];
  v8f accO[4];
#pragma unroll
  for (int r = 0; r < 8; ++r) { m_i[r] = -1e30f; l_i[r] = 0.0f; }
#pragma unroll
  for (int j = 0; j < 4; ++j) accO[j] = v8f_zero();

  __bf16* pb = &lP[wid * 16 * LDSTR];

  for (int kc = 0; kc < SEQ; kc += 32) {
    __syncthreads();
    // stage V chunk transposed: 32 keys x 64 dims -> lVt[d][key]
#pragma unroll
    for (int i = 0; i < 8; ++i) {
      int idx = tid + i * 256;
      int key = idx >> 6, d = idx & 63;
      lVt[d * LDSTR + key] = vw[base + (size_t)(kc + key) * HD + d];
    }
    __syncthreads();

    // scores: 16 queries x 32 keys, K-dim 64 (2 chained WMMAs per 16-key tile)
    v8f sc[2];
    sc[0] = v8f_zero(); sc[1] = v8f_zero();
#pragma unroll
    for (int t = 0; t < 2; ++t) {
#pragma unroll
      for (int ks = 0; ks < 2; ++ks) {
        FragBF fk;
        const __bf16* kp =
            kw + base + (size_t)(kc + ks * 16 + ln) * HD + t * 32 + hi * 8;
        fk.q[0] = *(const uint4*)(kp);
        fk.q[1] = *(const uint4*)(kp + 16);
        sc[ks] = wmma_bf16(fq[t].v, fk.v, sc[ks]);
      }
    }

    // online softmax: row m = r + 8*hi lives in 16 lanes -> xor reductions
#pragma unroll
    for (int r = 0; r < 8; ++r) {
      float s0 = sc[0][r] * 0.125f;   // 1/sqrt(64)
      float s1 = sc[1][r] * 0.125f;
      float mx = fmaxf(s0, s1);
#pragma unroll
      for (int m = 1; m < 16; m <<= 1) mx = fmaxf(mx, __shfl_xor(mx, m));
      float mnew = fmaxf(m_i[r], mx);
      float corr = __expf(m_i[r] - mnew);
      float p0 = __expf(s0 - mnew);
      float p1 = __expf(s1 - mnew);
      float rs = p0 + p1;
#pragma unroll
      for (int m = 1; m < 16; m <<= 1) rs += __shfl_xor(rs, m);
      l_i[r] = l_i[r] * corr + rs;
      m_i[r] = mnew;
#pragma unroll
      for (int j = 0; j < 4; ++j) accO[j][r] *= corr;
      // stash P (bf16) into per-wave LDS in A-fragment source layout
      pb[(r + 8 * hi) * LDSTR + ln]      = (__bf16)p0;
      pb[(r + 8 * hi) * LDSTR + 16 + ln] = (__bf16)p1;
    }

    // reload P as a 16x32 A-fragment (same-wave LDS RAW, hw waits inserted)
    FragBF fp;
    {
      const __bf16* pp = pb + ln * LDSTR + hi * 8;
      fp.q[0] = *(const uint4*)(pp);
      fp.q[1] = *(const uint4*)(pp + 16);
    }

    // P * V over 4 d-slices of 16
#pragma unroll
    for (int j = 0; j < 4; ++j) {
      FragBF fv;
      const __bf16* vp = &lVt[(j * 16 + ln) * LDSTR + hi * 8];
      fv.q[0] = *(const uint4*)(vp);
      fv.q[1] = *(const uint4*)(vp + 16);
      accO[j] = wmma_bf16(fp.v, fv.v, accO[j]);
    }
  }

  // normalize and store f32 output [b, n, h*64 + d]
#pragma unroll
  for (int r = 0; r < 8; ++r) {
    float linv = 1.0f / l_i[r];
    int n = q0 + r + 8 * hi;
    float* op = out + ((size_t)b * SEQ + n) * DIM + head * HD;
#pragma unroll
    for (int j = 0; j < 4; ++j)
      op[j * 16 + ln] = accO[j][r] * linv;
  }
}

// ---------------------------------------------------------------------------
extern "C" void kernel_launch(void* const* d_in, const int* in_sizes, int n_in,
                              void* d_out, int out_size, void* d_ws, size_t ws_size,
                              hipStream_t stream) {
  (void)in_sizes; (void)n_in; (void)out_size; (void)ws_size;
  const float* x    = (const float*)d_in[0];
  const float* wqkv = (const float*)d_in[1];
  const float* bqkv = (const float*)d_in[2];
  float* out = (float*)d_out;

  const size_t per = (size_t)BATCH * NHEADS * SEQ * HD;  // 8,388,608 elems
  __bf16* qw = (__bf16*)d_ws;
  __bf16* kw = qw + per;
  __bf16* vw = kw + per;
  __bf16* xb = vw + per;                       // [8192][1024] bf16
  __bf16* wt = xb + (size_t)MTOT * DIM;        // [3072][1024] bf16 (W^T)

  dim3 blk(256);
  cvt_x_kernel<<<dim3((MTOT * DIM) / (256 * 8)), blk, 0, stream>>>(x, xb);
  cvt_w_kernel<<<dim3(NOUT / 64, DIM / 64), blk, 0, stream>>>(wqkv, wt);

  dim3 g1((MTOT / 128) * (NOUT / 128));        // 64 * 24 = 1536 blocks
  qkv_rope_kernel<<<g1, blk, 0, stream>>>(xb, wt, bqkv, qw, kw, vw);

  dim3 g2(BATCH * NHEADS * (SEQ / 128));       // 1024 blocks
  attn_kernel<<<g2, blk, 0, stream>>>(qw, kw, vw, out);
}
